// Self_Attn_82849919140014
// MI455X (gfx1250) — matrix operand order repaired and checked by reading, not compiled, not run on hardware
//
#include <hip/hip_runtime.h>
#include <math.h>

// ---------------- problem constants ----------------
#define B_SZ  4
#define C_DIM 256
#define CQ    32
#define N_SP  3136          // 56*56
#define NJT   49            // N_SP / 64 (projection i-tiles)
#define NJB   98            // N_SP / 32 (attention j-blocks)

// ---------------- workspace layout (bytes) ----------------
static const size_t SCAL_OFF = 0;                                   // 1 float: log2e / norm_value
static const size_t BC_OFF   = 256;                                 // 320 floats fused bias
static const size_t WC_OFF   = 2048;                                // bf16 [320][256] fused weights
static const size_t Q_OFF    = 262144;                              // bf16 q^T [B][N][32]
static const size_t K_OFF    = Q_OFF + (size_t)B_SZ * N_SP * CQ * 2;    // bf16 k^T [B][N][32]
static const size_t V_OFF    = K_OFF + (size_t)B_SZ * N_SP * CQ * 2;    // bf16 v   [B][256][N]
// total ws use ~8.3 MB

// ---------------- types ----------------
typedef __attribute__((ext_vector_type(16))) __bf16        v16bf;
typedef __attribute__((ext_vector_type(8)))  float         v8f;
typedef __attribute__((ext_vector_type(4)))  unsigned int  u32x4;

// ---------------- helpers ----------------
__device__ __forceinline__ unsigned short f2bf(float f) {
  unsigned int u = __float_as_uint(f);
  u += 0x7FFFu + ((u >> 16) & 1u);          // round-to-nearest-even
  return (unsigned short)(u >> 16);
}

// two 16B chunks -> one 16-element bf16 operand (b128 loads)
__device__ __forceinline__ v16bf ldpair(const unsigned short* p0, const unsigned short* p1) {
  union { u32x4 u[2]; v16bf v; } x;
  x.u[0] = *(const u32x4*)p0;
  x.u[1] = *(const u32x4*)p1;
  return x.v;
}

__device__ __forceinline__ v8f wmma_bf16(v16bf a, v16bf b, v8f c) {
  return __builtin_amdgcn_wmma_f32_16x16x32_bf16(false, a, false, b, (short)0, c, false, false);
}

__device__ __forceinline__ float redmax16(float v) {
  v = fmaxf(v, __shfl_xor(v, 1, 16));
  v = fmaxf(v, __shfl_xor(v, 2, 16));
  v = fmaxf(v, __shfl_xor(v, 4, 16));
  v = fmaxf(v, __shfl_xor(v, 8, 16));
  return v;
}
__device__ __forceinline__ float redsum16(float v) {
  v += __shfl_xor(v, 1, 16);
  v += __shfl_xor(v, 2, 16);
  v += __shfl_xor(v, 4, 16);
  v += __shfl_xor(v, 8, 16);
  return v;
}

// ---- CDNA5 async global->LDS copy (ASYNCcnt path), per-lane 16B ----
// LDS address taken from the lds_off VGPR; low 32 bits of a generic shared pointer
// are the LDS byte offset (ISA 10.2 aperture rules).
__device__ __forceinline__ void async_g2l_b128(const unsigned short* lds_dst,
                                               const unsigned short* gsrc) {
  unsigned lds_off = (unsigned)(unsigned long long)lds_dst;
  asm volatile("global_load_async_to_lds_b128 %0, %1, off"
               :: "v"(lds_off), "v"(gsrc)
               : "memory");
}
__device__ __forceinline__ void wait_async0() {
  asm volatile("s_wait_asynccnt 0" ::: "memory");
}

// ======================================================================
// Kernel 1: softmax scale from weight norms.
// norm_value = ||Wq||_F * max( max_c ||Wk[:,c]||, max_c ||Wv[:,c]|| )
// store log2(e)/norm_value  (softmax computed in base 2 -> v_exp_f32)
// ======================================================================
__global__ void norm_kernel(const float* __restrict__ Wq, const float* __restrict__ Wk,
                            const float* __restrict__ Wv, char* __restrict__ ws) {
  __shared__ float red[256];
  int t = threadIdx.x;
  float fr = 0.f;
  for (int k = t; k < CQ * C_DIM; k += 256) fr += Wq[k] * Wq[k];
  red[t] = fr; __syncthreads();
  for (int s = 128; s > 0; s >>= 1) { if (t < s) red[t] += red[t + s]; __syncthreads(); }
  float u2 = red[0];
  __syncthreads();

  float sk = 0.f;
  for (int o = 0; o < CQ; ++o)    { float w = Wk[o * C_DIM + t]; sk += w * w; }
  float sv = 0.f;
  for (int o = 0; o < C_DIM; ++o) { float w = Wv[o * C_DIM + t]; sv += w * w; }
  red[t] = fmaxf(sk, sv); __syncthreads();
  for (int s = 128; s > 0; s >>= 1) { if (t < s) red[t] = fmaxf(red[t], red[t + s]); __syncthreads(); }
  if (t == 0) {
    float norm = sqrtf(u2) * sqrtf(red[0]);
    ((float*)(ws + SCAL_OFF))[0] = 1.4426950408889634f / norm;
  }
}

// ======================================================================
// Kernel 2: fuse Wq/Wk/Wv -> bf16 [320][256], biases -> f32 [320]
// ======================================================================
__global__ void cvt_w_kernel(const float* __restrict__ Wq, const float* __restrict__ Wk,
                             const float* __restrict__ Wv, const float* __restrict__ bq,
                             const float* __restrict__ bk, const float* __restrict__ bv,
                             char* __restrict__ ws) {
  int row = blockIdx.x, c = threadIdx.x;
  float w, bias;
  if (row < 32)      { w = Wq[row * C_DIM + c];        bias = bq[row]; }
  else if (row < 64) { w = Wk[(row - 32) * C_DIM + c]; bias = bk[row - 32]; }
  else               { w = Wv[(row - 64) * C_DIM + c]; bias = bv[row - 64]; }
  ((unsigned short*)(ws + WC_OFF))[row * C_DIM + c] = f2bf(w);
  if (c == 0) ((float*)(ws + BC_OFF))[row] = bias;
}

// ======================================================================
// Kernel 3: projections q,k,v via bf16 WMMA.
// Block = (b, 64-row i-tile); wave w owns 16 rows, all 20 output tiles.
// ======================================================================
__global__ __launch_bounds__(128, 1)
void proj_kernel(const float* __restrict__ x, char* __restrict__ ws) {
  __shared__ __align__(16) unsigned short xT[64 * 264];   // [i][c] bf16, row pad 264

  const int tid  = threadIdx.x;
  const int wave = tid >> 5;
  const int lane = tid & 31;
  const int col  = lane & 15;
  const int sel  = lane >> 4;

  const int b  = blockIdx.x / NJT;
  const int i0 = (blockIdx.x % NJT) * 64;

  // stage x tile [256 c][64 i] transposed to bf16 [i][c]
  for (int idx = tid; idx < C_DIM * 64; idx += 128) {
    int c = idx >> 6, ii = idx & 63;
    float v = x[((size_t)b * C_DIM + c) * N_SP + i0 + ii];
    xT[ii * 264 + c] = f2bf(v);
  }
  __syncthreads();

  const unsigned short* WC = (const unsigned short*)(ws + WC_OFF);
  const float*          bc = (const float*)(ws + BC_OFF);
  unsigned short* qhat = (unsigned short*)(ws + Q_OFF);
  unsigned short* khat = (unsigned short*)(ws + K_OFF);
  unsigned short* vhat = (unsigned short*)(ws + V_OFF);

  const int iloc = wave * 16;
  v8f vzero = {0.f, 0.f, 0.f, 0.f, 0.f, 0.f, 0.f, 0.f};

  for (int t = 0; t < 20; ++t) {               // 20 o-tiles: q(2) k(2) v(16)
    const int o0 = t * 16;
    v8f acc = vzero;
#pragma unroll
    for (int kc = 0; kc < 8; ++kc) {           // K-loop over C=256
      const unsigned short* wr = WC + (size_t)(o0 + col) * C_DIM + kc * 32;
      v16bf wa = ldpair(wr + 16 * sel, wr + 16 * sel + 8);                // A: W rows
      const unsigned short* xr = &xT[(iloc + col) * 264 + kc * 32 + 16 * sel];
      v16bf xb = ldpair(xr, xr + 8);                                      // B: x cols
      acc = wmma_bf16(wa, xb, acc);
    }
#pragma unroll
    for (int r = 0; r < 8; ++r) {              // D: row o, col i
      int o = o0 + r + 8 * sel;
      int i = i0 + iloc + col;
      unsigned short h = f2bf(acc[r] + bc[o]);
      if (o < 32)      qhat[((size_t)b * N_SP + i) * CQ + o]              = h;
      else if (o < 64) khat[((size_t)b * N_SP + i) * CQ + (o - 32)]       = h;
      else             vhat[((size_t)b * C_DIM + (o - 64)) * N_SP + i]    = h;
    }
  }
}

// ======================================================================
// Kernel 4: fused flash attention + epilogue, double-buffered async LDS.
// Block = (b, 64-row i-tile); wave owns 16 i-rows, acc [16i x 256c] VGPRs.
// j-block = 32 columns.  K tile [32j x 32c], V tile [256c x 32j] staged
// via global_load_async_to_lds_b128 (ASYNCcnt) with 2 buffers: fetch of
// block j+1 overlaps WMMA compute on block j.
// Pass A: row max + row sum (online).  Pass B: P = exp2(S - m)/l, O += P*V.
// ======================================================================
__device__ __forceinline__ void fetch_k(unsigned short* dst, const unsigned short* khat_b,
                                        int j0, int tid) {
  // 32 rows x 64B ; 128 lanes x 16B = one async instruction per wave
  int r = tid >> 2, p = tid & 3;
  async_g2l_b128(&dst[r * 40 + p * 8], khat_b + (size_t)(j0 + r) * CQ + p * 8);
}

__device__ __forceinline__ void fetch_v(unsigned short* dst, const unsigned short* vhat_b,
                                        int j0, int tid) {
  // 256 rows x 64B ; 8 async instructions per wave
#pragma unroll
  for (int k = 0; k < 8; ++k) {
    int idx = tid + k * 128;
    int c = idx >> 2, p = idx & 3;
    async_g2l_b128(&dst[c * 40 + p * 8], vhat_b + (size_t)c * N_SP + j0 + p * 8);
  }
}

__global__ __launch_bounds__(128, 1)
void attn_kernel(const float* __restrict__ x, const float* __restrict__ gamma_p,
                 const char* __restrict__ ws, float* __restrict__ out, float inv_bound) {
  __shared__ __align__(16) unsigned short ktil[2][32 * 40];    // K^T tile [j][c], row pad 40
  __shared__ __align__(16) unsigned short vtil[2][C_DIM * 40]; // V tile [c][j], row pad 40
  __shared__ __align__(16) unsigned short psc[4][16 * 40];     // per-wave P scratch [16][40]

  const int tid  = threadIdx.x;
  const int wave = tid >> 5;
  const int lane = tid & 31;
  const int col  = lane & 15;
  const int sel  = lane >> 4;

  const int b  = blockIdx.x / NJT;
  const int i0 = (blockIdx.x % NJT) * 64 + wave * 16;

  const float scale_s = *(const float*)(ws + SCAL_OFF);
  const unsigned short* qhat = (const unsigned short*)(ws + Q_OFF);
  const unsigned short* khat_b = (const unsigned short*)(ws + K_OFF) + (size_t)b * N_SP * CQ;
  const unsigned short* vhat_b = (const unsigned short*)(ws + V_OFF) + (size_t)b * C_DIM * N_SP;

  // Q A-tile (16 i x 32 c) lives in registers for the whole kernel
  const unsigned short* qrow = qhat + ((size_t)b * N_SP + i0 + col) * CQ;
  v16bf qa = ldpair(qrow + 16 * sel, qrow + 16 * sel + 8);

  v8f vzero = {0.f, 0.f, 0.f, 0.f, 0.f, 0.f, 0.f, 0.f};

  float m[8], l[8];
#pragma unroll
  for (int r = 0; r < 8; ++r) { m[r] = -1e30f; l[r] = 0.f; }

  // ---------------- Pass A: softmax statistics ----------------
  fetch_k(ktil[0], khat_b, 0, tid);
  for (int jb = 0; jb < NJB; ++jb) {
    wait_async0();
    __syncthreads();                              // buffer jb&1 ready for everyone
    if (jb + 1 < NJB) fetch_k(ktil[(jb + 1) & 1], khat_b, (jb + 1) * 32, tid);
    const unsigned short* kt = ktil[jb & 1];

    float st[2][8];
#pragma unroll
    for (int jt = 0; jt < 2; ++jt) {
      const unsigned short* kr = &kt[(jt * 16 + col) * 40 + 16 * sel];
      v16bf kb = ldpair(kr, kr + 8);
      v8f s = wmma_bf16(qa, kb, vzero);
#pragma unroll
      for (int r = 0; r < 8; ++r) st[jt][r] = s[r] * scale_s;
    }
#pragma unroll
    for (int r = 0; r < 8; ++r) {
      float v0 = fmaxf(st[0][r], st[1][r]);
      float M  = redmax16(v0);
      float mn = fmaxf(m[r], M);
      float e  = exp2f(st[0][r] - mn) + exp2f(st[1][r] - mn);
      float S  = redsum16(e);
      l[r] = l[r] * exp2f(m[r] - mn) + S;
      m[r] = mn;
    }
  }
  __syncthreads();

  float invl[8];
#pragma unroll
  for (int r = 0; r < 8; ++r) invl[r] = 1.0f / l[r];

  // ---------------- Pass B: O = P * V^T ----------------
  v8f acc[16];
#pragma unroll
  for (int ct = 0; ct < 16; ++ct) acc[ct] = vzero;

  fetch_k(ktil[0], khat_b, 0, tid);
  fetch_v(vtil[0], vhat_b, 0, tid);
  for (int jb = 0; jb < NJB; ++jb) {
    wait_async0();
    __syncthreads();                              // buffers jb&1 ready
    if (jb + 1 < NJB) {
      fetch_k(ktil[(jb + 1) & 1], khat_b, (jb + 1) * 32, tid);
      fetch_v(vtil[(jb + 1) & 1], vhat_b, (jb + 1) * 32, tid);
    }
    const unsigned short* kt = ktil[jb & 1];
    const unsigned short* vt = vtil[jb & 1];

    unsigned short* pw = psc[wave];
#pragma unroll
    for (int jt = 0; jt < 2; ++jt) {
      const unsigned short* kr = &kt[(jt * 16 + col) * 40 + 16 * sel];
      v16bf kb = ldpair(kr, kr + 8);
      v8f s = wmma_bf16(qa, kb, vzero);
#pragma unroll
      for (int r = 0; r < 8; ++r) {
        float p = exp2f(s[r] * scale_s - m[r]) * invl[r];
        pw[(r + 8 * sel) * 40 + jt * 16 + col] = f2bf(p);   // C-layout -> [row][col]
      }
    }
    // reload P in A-matrix layout (16 i x 32 j)
    const unsigned short* pr = &psc[wave][col * 40];
    v16bf pa = ldpair(pr + 16 * sel, pr + 16 * sel + 8);
#pragma unroll
    for (int ct = 0; ct < 16; ++ct) {
      const unsigned short* vr = &vt[(ct * 16 + col) * 40 + 16 * sel];
      v16bf vb = ldpair(vr, vr + 8);
      acc[ct] = wmma_bf16(pa, vb, acc[ct]);
    }
  }

  // ---------------- epilogue: gamma * O / bound + x ----------------
  const float g = gamma_p[0];
#pragma unroll
  for (int ct = 0; ct < 16; ++ct) {
#pragma unroll
    for (int r = 0; r < 8; ++r) {
      int i = i0 + r + 8 * sel;                 // D rows = i
      int c = ct * 16 + col;                    // D cols = channel
      size_t off = ((size_t)b * C_DIM + c) * N_SP + i;
      out[off] = g * (acc[ct][r] * inv_bound) + x[off];
    }
  }
}

// ======================================================================
extern "C" void kernel_launch(void* const* d_in, const int* in_sizes, int n_in,
                              void* d_out, int out_size, void* d_ws, size_t ws_size,
                              hipStream_t stream) {
  const float* x     = (const float*)d_in[0];
  const float* Wq    = (const float*)d_in[1];
  const float* bq    = (const float*)d_in[2];
  const float* Wk    = (const float*)d_in[3];
  const float* bk    = (const float*)d_in[4];
  const float* Wv    = (const float*)d_in[5];
  const float* bv    = (const float*)d_in[6];
  const float* gamma = (const float*)d_in[7];
  float* out = (float*)d_out;
  char*  ws  = (char*)d_ws;

  // lipschitz_bound = exp(sqrt(3)) * sqrt(N/C) + 2*sqrt(6)
  const float inv_bound =
      1.0f / (expf(sqrtf(3.0f)) * sqrtf((float)N_SP / (float)C_DIM) + 2.0f * sqrtf(6.0f));

  norm_kernel <<<1,   256, 0, stream>>>(Wq, Wk, Wv, ws);
  cvt_w_kernel<<<320, 256, 0, stream>>>(Wq, Wk, Wv, bq, bk, bv, ws);
  proj_kernel <<<B_SZ * NJT, 128, 0, stream>>>(x, ws);
  attn_kernel <<<B_SZ * NJT, 128, 0, stream>>>(x, gamma, ws, out, inv_bound);
}
